// MambaBackbone_33569464385683
// MI455X (gfx1250) — compile-verified, matrix operand order checked by reading, not run on hardware
//
#include <hip/hip_runtime.h>

// ---------------- problem constants ----------------
#define B_SZ     2
#define L_TOK    576
#define M_TOK    (B_SZ * L_TOK)      // 1152 rows in every GEMM
#define D_MODEL  1024
#define D_INNER  2048
#define D_STATE  16
#define D_TR     64
#define N_XPROJ  96                   // valid rows of xproj weight
#define N_XPAD   128                  // padded to a full 128-wide GEMM tile
#define N_LAYERS 24
#define K_PATCH  (3 * 16 * 16)        // 768
#define IMG_ELEMS (B_SZ * 3 * 384 * 384)

typedef __attribute__((ext_vector_type(16))) __bf16 v16bf;
typedef __attribute__((ext_vector_type(2)))  __bf16 v2bf;
typedef __attribute__((ext_vector_type(8)))  float  v8f;

// fp32 -> bf16 via native conversion
__device__ __forceinline__ unsigned short f2bf(float f) {
  union { __bf16 h; unsigned short u; } cv;
  cv.h = (__bf16)f;
  return cv.u;
}

__device__ __forceinline__ unsigned int f2bf2(float a, float b) {
  union { v2bf v; unsigned int u; } cv;
  cv.v[0] = (__bf16)a;
  cv.v[1] = (__bf16)b;
  return cv.u;
}

// fast softplus: max(x,0) + log(1+exp(-|x|)) with hw transcendentals
__device__ __forceinline__ float softplus_fast(float x) {
  return fmaxf(x, 0.f) + __logf(1.f + __expf(-fabsf(x)));
}

// Load one 16x32 bf16 fragment. Row-major source, LD = K (elements).
// lane&15 -> row, lane>>4 -> which 16-wide K half. The same mapping is used
// for the A operand (lane=M) and the W operand (lane=N); the operand layouts
// are symmetric, so the consistent K permutation cancels in the product.
template <int LD>
__device__ __forceinline__ v16bf load_frag(const unsigned short* __restrict__ p,
                                           int row0, int k0) {
  const int lane = threadIdx.x & 31;
  const int r = row0 + (lane & 15);
  const int k = k0 + ((lane >> 4) << 4);
  return *(const v16bf*)(p + (size_t)r * LD + k);
}

// ---------------- WMMA GEMM (N multiple of 128, branch-free) ----------------
// C[M,N] = A[M,K] * W[N,K]^T.  Wave tile 64x64 (16 WMMAs / K-step),
// block = 4 waves in 2x2 -> 128x128 tile.  grid = (N/128, M/128).
// EPI: 0 = plain store, 1 = fast-softplus(x + bias[n]), 2 = C += result
// N, K compile-time: strides fold into shifts / instruction offsets.
template <int EPI, int N, int K>
__global__ __launch_bounds__(128) void gemm_big_k(
    const unsigned short* __restrict__ A, const unsigned short* __restrict__ W,
    float* __restrict__ C, const float* __restrict__ bias) {
  const int lane = threadIdx.x & 31;
  const int wave = threadIdx.x >> 5;
  const int m0 = blockIdx.y * 128 + (wave >> 1) * 64;
  const int n0 = blockIdx.x * 128 + (wave & 1) * 64;

  v8f acc[4][4];
#pragma unroll
  for (int i = 0; i < 4; ++i)
#pragma unroll
    for (int j = 0; j < 4; ++j)
#pragma unroll
      for (int t = 0; t < 8; ++t) acc[i][j][t] = 0.f;

#pragma unroll 2
  for (int k0 = 0; k0 < K; k0 += 32) {
    v16bf a[4], b[4];
#pragma unroll
    for (int i = 0; i < 4; ++i) a[i] = load_frag<K>(A, m0 + i * 16, k0);
#pragma unroll
    for (int j = 0; j < 4; ++j) b[j] = load_frag<K>(W, n0 + j * 16, k0);
#pragma unroll
    for (int i = 0; i < 4; ++i)
#pragma unroll
      for (int j = 0; j < 4; ++j)
        acc[i][j] = __builtin_amdgcn_wmma_f32_16x16x32_bf16(
            false, a[i], false, b[j], (short)0, acc[i][j], false, false);
  }

  // C/D layout: lane&15 -> column, lane>>4 selects rows {0..7} vs {8..15}
  const int ncol = lane & 15;
  const int rofs = (lane >> 4) << 3;
#pragma unroll
  for (int i = 0; i < 4; ++i) {
    const int mrow = m0 + i * 16 + rofs;
#pragma unroll
    for (int j = 0; j < 4; ++j) {
      const int n = n0 + j * 16 + ncol;
      float* cp = C + (size_t)mrow * N + n;
#pragma unroll
      for (int t = 0; t < 8; ++t) {
        float v = acc[i][j][t];
        if (EPI == 1) v = softplus_fast(v + bias[n]);
        if (EPI == 2) v += cp[(size_t)t * N];
        cp[(size_t)t * N] = v;
      }
    }
  }
}

// ---------------- patch embed (implicit-im2col WMMA, bf16 image) -----------
// tok[m, n] = sum_k im2col(xb)[m,k] * patch_w[n,k] + patch_b[n] + pos[m%L, n]
// K index k = c*256 + ph*16 + pw; each 16-aligned 16-chunk of K is one fixed
// (c, ph) with pw=0..15 -> 16 contiguous bf16 in the image row (32B aligned).
__global__ __launch_bounds__(128) void patch_embed_k(
    const unsigned short* __restrict__ xb, const unsigned short* __restrict__ Wp,
    const float* __restrict__ pb, const float* __restrict__ pos,
    float* __restrict__ tok) {
  const int lane = threadIdx.x & 31;
  const int m0 = blockIdx.y * 64 + (threadIdx.x >> 5) * 16;
  const int n0 = blockIdx.x * 64;

  const int mA = m0 + (lane & 15);
  const int bA = mA / L_TOK;
  const int tA = mA % L_TOK;
  const int hh = tA / 24, ww = tA % 24;
  const int koff = (lane >> 4) << 4;

  v8f acc[4];
#pragma unroll
  for (int j = 0; j < 4; ++j)
#pragma unroll
    for (int t = 0; t < 8; ++t) acc[j][t] = 0.f;

#pragma unroll 2
  for (int k0 = 0; k0 < K_PATCH; k0 += 32) {
    const int k = k0 + koff;
    const int c = k >> 8;
    const int ph = (k >> 4) & 15;
    const v16bf a = *(const v16bf*)(
        xb + (((size_t)(bA * 3 + c) * 384 + hh * 16 + ph) * 384 + ww * 16));
#pragma unroll
    for (int j = 0; j < 4; ++j) {
      v16bf b = load_frag<K_PATCH>(Wp, n0 + j * 16, k0);
      acc[j] = __builtin_amdgcn_wmma_f32_16x16x32_bf16(
          false, a, false, b, (short)0, acc[j], false, false);
    }
  }

  const int ncol = lane & 15;
  const int mrow = m0 + ((lane >> 4) << 3);
  // tiles are 16 rows and 576 % 16 == 0, so a tile never crosses a batch
  // boundary: hoist the expensive % L_TOK out of the store loop.
  const int tbase = mrow % L_TOK;
#pragma unroll
  for (int j = 0; j < 4; ++j) {
    const int n = n0 + j * 16 + ncol;
    const float pbn = pb[n];
#pragma unroll
    for (int t = 0; t < 8; ++t) {
      tok[(size_t)(mrow + t) * D_MODEL + n] =
          acc[j][t] + pbn + pos[(size_t)(tbase + t) * D_MODEL + n];
    }
  }
}

// ---------------- LayerNorm -> bf16 ----------------
__global__ __launch_bounds__(256) void layernorm_k(
    const float* __restrict__ x, const float* __restrict__ g,
    const float* __restrict__ b, unsigned int* __restrict__ u) {
  __shared__ float s1[256], s2[256];
  const int row = blockIdx.x;
  const float* xr = x + (size_t)row * D_MODEL;
  float a0 = 0.f, a1 = 0.f;
  for (int i = threadIdx.x; i < D_MODEL; i += 256) {
    const float v = xr[i];
    a0 += v; a1 += v * v;
  }
  s1[threadIdx.x] = a0; s2[threadIdx.x] = a1;
  __syncthreads();
  for (int st = 128; st > 0; st >>= 1) {
    if (threadIdx.x < st) {
      s1[threadIdx.x] += s1[threadIdx.x + st];
      s2[threadIdx.x] += s2[threadIdx.x + st];
    }
    __syncthreads();
  }
  const float mu = s1[0] * (1.f / D_MODEL);
  const float var = s2[0] * (1.f / D_MODEL) - mu * mu;
  const float inv = rsqrtf(var + 1e-5f);
  for (int i = threadIdx.x; i < D_MODEL / 2; i += 256) {
    const float v0 = (xr[2 * i] - mu) * inv * g[2 * i] + b[2 * i];
    const float v1 = (xr[2 * i + 1] - mu) * inv * g[2 * i + 1] + b[2 * i + 1];
    u[(size_t)row * (D_MODEL / 2) + i] = f2bf2(v0, v1);
  }
}

// ---------------- causal depthwise conv (DC=4) + SiLU ----------------
// xi = xz[:, 0:DI]; xc = silu(sum_k wc[e,k] * xi[l-3+k, e] + bc[e])
__global__ __launch_bounds__(256) void conv_silu_k(
    const float* __restrict__ xz, const float* __restrict__ wc,
    const float* __restrict__ bc, float* __restrict__ xcf,
    unsigned int* __restrict__ xcb) {
  const int pidx = blockIdx.x * 256 + threadIdx.x;  // pair index
  if (pidx >= M_TOK * D_INNER / 2) return;
  const int r = pidx / (D_INNER / 2);  // b*L + l
  const int e = (pidx - r * (D_INNER / 2)) * 2;
  const int l = r % L_TOK;
  float sv[2];
#pragma unroll
  for (int q = 0; q < 2; ++q) {
    const int ee = e + q;
    const size_t base = (size_t)r * (2 * D_INNER) + ee;
    float acc = bc[ee] + wc[ee * 4 + 3] * xz[base];
    if (l >= 1) acc += wc[ee * 4 + 2] * xz[base - (2 * D_INNER)];
    if (l >= 2) acc += wc[ee * 4 + 1] * xz[base - 2 * (2 * D_INNER)];
    if (l >= 3) acc += wc[ee * 4 + 0] * xz[base - 3 * (2 * D_INNER)];
    sv[q] = acc / (1.f + __expf(-acc));  // silu
    xcf[(size_t)r * D_INNER + ee] = sv[q];
  }
  xcb[pidx] = f2bf2(sv[0], sv[1]);
}

// ---------------- selective scan + gate ----------------
// thread = (batch,channel,state); 16-lane shfl reduction over the state dim.
// dbc has leading dimension N_XPAD (=128); B at col 64+s, C at col 80+s.
__global__ __launch_bounds__(256) void scan_k(
    const float* __restrict__ dtv, const float* __restrict__ xcf,
    const float* __restrict__ dbc, const float* __restrict__ xz,
    const float* __restrict__ Alog, const float* __restrict__ Dp,
    unsigned short* __restrict__ yb) {
  const int tid = blockIdx.x * 256 + threadIdx.x;  // B*DI*DS = 65536
  const int s  = tid & 15;
  const int ch = tid >> 4;             // 0..B*DI-1
  const int e  = ch & (D_INNER - 1);
  const int b  = ch >> 11;
  const float Aval = -__expf(Alog[e * D_STATE + s]);
  const float De = Dp[e];
  float h = 0.f;
  for (int l = 0; l < L_TOK; ++l) {
    const size_t r = (size_t)(b * L_TOK + l);
    const float dt = dtv[r * D_INNER + e];
    const float xc = xcf[r * D_INNER + e];
    const float Bm = dbc[r * N_XPAD + D_TR + s];
    const float Cm = dbc[r * N_XPAD + D_TR + D_STATE + s];
    h = __expf(dt * Aval) * h + dt * xc * Bm;
    float p = h * Cm;
    p += __shfl_xor(p, 1, 16);
    p += __shfl_xor(p, 2, 16);
    p += __shfl_xor(p, 4, 16);
    p += __shfl_xor(p, 8, 16);
    if (s == 0) {
      const float z = xz[r * (2 * D_INNER) + D_INNER + e];
      const float y = (p + De * xc) * (z / (1.f + __expf(-z)));
      yb[r * D_INNER + e] = f2bf(y);
    }
  }
}

// ---------------- converters ----------------
__global__ void cvt_bf16_k(const float* __restrict__ s,
                           unsigned int* __restrict__ d, int npairs) {
  const int i = blockIdx.x * 256 + threadIdx.x;
  if (i < npairs) d[i] = f2bf2(s[2 * i], s[2 * i + 1]);
}

// xproj weight: convert 96 valid rows, zero-fill rows 96..127
__global__ void cvt_xp_k(const float* __restrict__ s,
                         unsigned int* __restrict__ d) {
  const int i = blockIdx.x * 256 + threadIdx.x;  // pair index
  if (i >= N_XPAD * D_INNER / 2) return;
  const int r = i / (D_INNER / 2);
  const int c = (i - r * (D_INNER / 2)) * 2;
  d[i] = (r < N_XPROJ)
             ? f2bf2(s[(size_t)r * D_INNER + c], s[(size_t)r * D_INNER + c + 1])
             : 0u;
}

// dt rows: fp32 [1152, N_XPAD] cols 0..63 -> bf16 [1152, 64]
__global__ void cvt_dtr_k(const float* __restrict__ dbc,
                          unsigned int* __restrict__ dtr) {
  const int i = blockIdx.x * 256 + threadIdx.x;  // pair index
  if (i >= M_TOK * D_TR / 2) return;
  const int r = i >> 5, c = (i & 31) * 2;
  dtr[i] = f2bf2(dbc[(size_t)r * N_XPAD + c], dbc[(size_t)r * N_XPAD + c + 1]);
}

// ---------------- host orchestration ----------------
extern "C" void kernel_launch(void* const* d_in, const int* in_sizes, int n_in,
                              void* d_out, int out_size, void* d_ws,
                              size_t ws_size, hipStream_t stream) {
  (void)in_sizes; (void)n_in; (void)out_size; (void)ws_size;
  const float* x       = (const float*)d_in[0];
  const float* patch_w = (const float*)d_in[1];
  const float* patch_b = (const float*)d_in[2];
  const float* pos     = (const float*)d_in[3];
  const float* ln_g    = (const float*)d_in[4];
  const float* ln_b    = (const float*)d_in[5];
  const float* in_w    = (const float*)d_in[6];
  const float* conv_w  = (const float*)d_in[7];
  const float* conv_b  = (const float*)d_in[8];
  const float* xproj_w = (const float*)d_in[9];
  const float* dt_w    = (const float*)d_in[10];
  const float* dt_b    = (const float*)d_in[11];
  const float* A_log   = (const float*)d_in[12];
  const float* D_par   = (const float*)d_in[13];
  const float* out_w   = (const float*)d_in[14];
  float* tok = (float*)d_out;  // resident [B*L, D] token buffer

  char* ws = (char*)d_ws;
  size_t off = 0;
  auto alloc = [&](size_t bytes) -> void* {
    off = (off + 255) & ~(size_t)255;
    void* p = ws + off;
    off += bytes;
    return p;
  };
  unsigned short* x_bf = (unsigned short*)alloc((size_t)IMG_ELEMS * 2);
  unsigned short* pw_b = (unsigned short*)alloc((size_t)D_MODEL * K_PATCH * 2);
  unsigned short* u_bf = (unsigned short*)alloc((size_t)M_TOK * D_MODEL * 2);
  float* xz            = (float*)alloc((size_t)M_TOK * 2 * D_INNER * 4);
  float* xcf           = (float*)alloc((size_t)M_TOK * D_INNER * 4);
  unsigned short* xcb  = (unsigned short*)alloc((size_t)M_TOK * D_INNER * 2);
  float* dbc           = (float*)alloc((size_t)M_TOK * N_XPAD * 4);
  unsigned short* dtrb = (unsigned short*)alloc((size_t)M_TOK * D_TR * 2);
  float* dtv           = (float*)alloc((size_t)M_TOK * D_INNER * 4);
  unsigned short* ybf  = (unsigned short*)alloc((size_t)M_TOK * D_INNER * 2);
  unsigned short* w_in = (unsigned short*)alloc((size_t)2 * D_INNER * D_MODEL * 2);
  unsigned short* w_xp = (unsigned short*)alloc((size_t)N_XPAD * D_INNER * 2);
  unsigned short* w_dt = (unsigned short*)alloc((size_t)D_INNER * D_TR * 2);
  unsigned short* w_ot = (unsigned short*)alloc((size_t)D_MODEL * D_INNER * 2);

  auto cvt = [&](const float* s, unsigned short* d, int n) {
    cvt_bf16_k<<<(n / 2 + 255) / 256, 256, 0, stream>>>(
        s, (unsigned int*)d, n / 2);
  };

  // ---- patch embed + pos embed (resize is identity: 24x24 -> 24x24) ----
  cvt(x, x_bf, IMG_ELEMS);
  cvt(patch_w, pw_b, D_MODEL * K_PATCH);
  patch_embed_k<<<dim3(D_MODEL / 64, M_TOK / 64), 128, 0, stream>>>(
      x_bf, pw_b, patch_b, pos, tok);

  for (int layer = 0; layer < N_LAYERS; ++layer) {
    cvt(in_w + (size_t)layer * 2 * D_INNER * D_MODEL, w_in, 2 * D_INNER * D_MODEL);
    cvt_xp_k<<<(N_XPAD * D_INNER / 2 + 255) / 256, 256, 0, stream>>>(
        xproj_w + (size_t)layer * N_XPROJ * D_INNER, (unsigned int*)w_xp);
    cvt(dt_w + (size_t)layer * D_INNER * D_TR, w_dt, D_INNER * D_TR);
    cvt(out_w + (size_t)layer * D_MODEL * D_INNER, w_ot, D_MODEL * D_INNER);

    layernorm_k<<<M_TOK, 256, 0, stream>>>(
        tok, ln_g + (size_t)layer * D_MODEL, ln_b + (size_t)layer * D_MODEL,
        (unsigned int*)u_bf);

    // xz = u @ Win^T   [1152, 4096]
    gemm_big_k<0, 2 * D_INNER, D_MODEL>
        <<<dim3(2 * D_INNER / 128, M_TOK / 128), 128, 0, stream>>>(
            u_bf, w_in, xz, nullptr);

    conv_silu_k<<<(M_TOK * D_INNER / 2 + 255) / 256, 256, 0, stream>>>(
        xz, conv_w + (size_t)layer * D_INNER * 4,
        conv_b + (size_t)layer * D_INNER, xcf, (unsigned int*)xcb);

    // dbc = xc @ Wxp_pad^T   [1152, 128] (cols 96..127 are zeros)
    gemm_big_k<0, N_XPAD, D_INNER>
        <<<dim3(N_XPAD / 128, M_TOK / 128), 128, 0, stream>>>(
            xcb, w_xp, dbc, nullptr);

    cvt_dtr_k<<<(M_TOK * D_TR / 2 + 255) / 256, 256, 0, stream>>>(
        dbc, (unsigned int*)dtrb);

    // dt = softplus(dt_r @ Wdt^T + bdt)   [1152, 2048]
    gemm_big_k<1, D_INNER, D_TR>
        <<<dim3(D_INNER / 128, M_TOK / 128), 128, 0, stream>>>(
            dtrb, w_dt, dtv, dt_b + (size_t)layer * D_INNER);

    scan_k<<<B_SZ * D_INNER * D_STATE / 256, 256, 0, stream>>>(
        dtv, xcf, dbc, xz, A_log + (size_t)layer * D_INNER * D_STATE,
        D_par + (size_t)layer * D_INNER, ybf);

    // tok += y @ Wout^T   (residual)
    gemm_big_k<2, D_MODEL, D_INNER>
        <<<dim3(D_MODEL / 128, M_TOK / 128), 128, 0, stream>>>(
            ybf, w_ot, tok, nullptr);
  }
}
